// Attention_3058016715034
// MI455X (gfx1250) — matrix operand order repaired and checked by reading, not compiled, not run on hardware
//
#include <hip/hip_runtime.h>

// ---------------- types ----------------
typedef __attribute__((ext_vector_type(16))) __bf16 v16bf;
typedef __attribute__((ext_vector_type(8)))  __bf16 bf8;
typedef __attribute__((ext_vector_type(8)))  float  v8f;
typedef __attribute__((ext_vector_type(4)))  unsigned int u32x4;
typedef __attribute__((ext_vector_type(8)))  int i32x8;
typedef __attribute__((ext_vector_type(4)))  int i32x4;

#define WMMA_BF16(a, b, c) \
  __builtin_amdgcn_wmma_f32_16x16x32_bf16(false, (a), false, (b), (short)0, (c), false, false)

__device__ __forceinline__ v16bf cat16(bf8 lo, bf8 hi) {
  return __builtin_shufflevector(lo, hi, 0,1,2,3,4,5,6,7,8,9,10,11,12,13,14,15);
}

// ---- 16-lane-row butterfly reductions (VALU permlane, no LDS) ----
#if __has_builtin(__builtin_amdgcn_permlane16)
__device__ __forceinline__ float pl16(float v, unsigned s0, unsigned s1) {
  int x = __builtin_bit_cast(int, v);
  int y = __builtin_amdgcn_permlane16(x, x, s0, s1, false, false);
  return __builtin_bit_cast(float, y);
}
__device__ __forceinline__ float rowmax16(float v) {
  v = fmaxf(v, pl16(v, 0x67452301u, 0xEFCDAB89u));  // xor 1
  v = fmaxf(v, pl16(v, 0x54761032u, 0xDCFE98BAu));  // xor 2
  v = fmaxf(v, pl16(v, 0x32107654u, 0xBA98FEDCu));  // xor 4
  v = fmaxf(v, pl16(v, 0xFEDCBA98u, 0x76543210u));  // xor 8
  return v;
}
__device__ __forceinline__ float rowsum16(float v) {
  v += pl16(v, 0x67452301u, 0xEFCDAB89u);
  v += pl16(v, 0x54761032u, 0xDCFE98BAu);
  v += pl16(v, 0x32107654u, 0xBA98FEDCu);
  v += pl16(v, 0xFEDCBA98u, 0x76543210u);
  return v;
}
#else
__device__ __forceinline__ float rowmax16(float v) {
  for (int msk = 1; msk < 16; msk <<= 1) v = fmaxf(v, __shfl_xor(v, msk, 32));
  return v;
}
__device__ __forceinline__ float rowsum16(float v) {
  for (int msk = 1; msk < 16; msk <<= 1) v += __shfl_xor(v, msk, 32);
  return v;
}
#endif

// ---- TDM: DMA a 64x64 bf16 tile (row stride in elements) into LDS with
//      16B row padding (LDS row pitch 144B == [64][72] bf16). ----
__device__ __forceinline__ void tdm_load_tile64x64_bf16(
    const void* gptr, unsigned lds_byte_addr, unsigned row_stride_elems) {
  unsigned long long ga = (unsigned long long)(uintptr_t)gptr;
  u32x4 g0;
  g0[0] = 1u;                                   // count=1, user descriptor
  g0[1] = lds_byte_addr;                        // lds_addr
  g0[2] = (unsigned)ga;                         // global_addr[31:0]
  g0[3] = (unsigned)(ga >> 32) | (2u << 30);    // global_addr[56:32] | type=2
  i32x8 g1;
  // data_size=2B | pad_enable | pad_interval=32dw(code4) | pad_amount=4dw(code3)
  g1[0] = (int)((1u << 16) | (1u << 20) | (4u << 22) | (3u << 25));
  g1[1] = (int)(64u << 16);                     // tensor_dim0[15:0]=64
  g1[2] = (int)(64u << 16);                     // tensor_dim1[15:0]=64
  g1[3] = (int)(64u << 16);                     // tile_dim0=64
  g1[4] = (int)64u;                             // tile_dim1=64, tile_dim2=0
  g1[5] = (int)row_stride_elems;                // tensor_dim0_stride[31:0]
  g1[6] = 0;
  g1[7] = 0;
  i32x4 z4 = {0, 0, 0, 0};
  i32x8 z8 = {0, 0, 0, 0, 0, 0, 0, 0};
  __builtin_amdgcn_tensor_load_to_lds(g0, g1, z4, z4, z8, 0);
}

// stage 16 contiguous elements of A (f32 or bf16) into LDS as bf16
__device__ __forceinline__ void load16_cvt(const float* __restrict__ src, __bf16* dst) {
#pragma unroll
  for (int v = 0; v < 4; ++v) {
    float4 t = ((const float4*)src)[v];
    dst[4*v+0] = (__bf16)t.x; dst[4*v+1] = (__bf16)t.y;
    dst[4*v+2] = (__bf16)t.z; dst[4*v+3] = (__bf16)t.w;
  }
}
__device__ __forceinline__ void load16_cvt(const __bf16* __restrict__ src, __bf16* dst) {
  *(bf8*)(dst)     = *(const bf8*)(src);
  *(bf8*)(dst + 8) = *(const bf8*)(src + 8);
}

// ---------------- GEMM: C[M,N] = A[M,K] @ W[K,N] + bias ----------------
// Tile 128x128, TK=32, 256 threads = 8 waves in 2(M) x 4(N); wave = 64x32.
template <typename AT, bool OUT_BF16>
__global__ __launch_bounds__(256) void gemm_bias_kernel(
    const AT* __restrict__ A, const float* __restrict__ W,
    const float* __restrict__ bias, void* __restrict__ Cout,
    int M, int N, int K) {
  __shared__ __bf16 As[128][40];   // [m][k], stride 40 for bank spread
  __shared__ __bf16 Bs[128][40];   // [n][k] (W transposed on stage-in)

  const int tid  = threadIdx.x;
  const int lane = tid & 31, wave = tid >> 5;
  const int wm = wave >> 2, wn = wave & 3;
  const int g = lane >> 4, m = lane & 15;
  const int row0 = blockIdx.y * 128;
  const int n0   = blockIdx.x * 128;

  v8f acc[4][2] = {};

  const int arow = tid >> 1, akb = (tid & 1) * 16;   // A stage: 2 thr/row x 16 elem
  const int bk   = tid >> 3, bnb = (tid & 7) * 16;   // W stage: 8 thr/row x 16 elem

  for (int k0 = 0; k0 < K; k0 += 32) {
    __syncthreads();
    load16_cvt(A + (size_t)(row0 + arow) * K + k0 + akb, &As[arow][akb]);
    {
      const float* wsrc = W + (size_t)(k0 + bk) * N + n0 + bnb;
#pragma unroll
      for (int v = 0; v < 4; ++v) {
        float4 t = ((const float4*)wsrc)[v];
        Bs[bnb + 4*v + 0][bk] = (__bf16)t.x;
        Bs[bnb + 4*v + 1][bk] = (__bf16)t.y;
        Bs[bnb + 4*v + 2][bk] = (__bf16)t.z;
        Bs[bnb + 4*v + 3][bk] = (__bf16)t.w;
      }
    }
    __syncthreads();

    v16bf af[4], bfr[2];
#pragma unroll
    for (int mi = 0; mi < 4; ++mi) {
      const __bf16* p = &As[wm*64 + mi*16 + m][0];
      af[mi] = cat16(*(const bf8*)(p + g*8), *(const bf8*)(p + 16 + g*8));
    }
#pragma unroll
    for (int ni = 0; ni < 2; ++ni) {
      const __bf16* p = &Bs[wn*32 + ni*16 + m][0];
      bfr[ni] = cat16(*(const bf8*)(p + g*8), *(const bf8*)(p + 16 + g*8));
    }
#pragma unroll
    for (int mi = 0; mi < 4; ++mi)
#pragma unroll
      for (int ni = 0; ni < 2; ++ni)
        acc[mi][ni] = WMMA_BF16(af[mi], bfr[ni], acc[mi][ni]);
  }

#pragma unroll
  for (int ni = 0; ni < 2; ++ni) {
    const int ncol = n0 + wn*32 + ni*16 + m;
    const float bv = bias[ncol];
#pragma unroll
    for (int mi = 0; mi < 4; ++mi) {
#pragma unroll
      for (int r = 0; r < 8; ++r) {
        const int row = row0 + wm*64 + mi*16 + g*8 + r;
        const float v = acc[mi][ni][r] + bv;
        if (OUT_BF16) ((__bf16*)Cout)[(size_t)row * N + ncol] = (__bf16)v;
        else          ((float*)Cout)[(size_t)row * N + ncol]  = v;
      }
    }
  }
}

// ---------------- Flash attention over qkv buffer [8192][2304] bf16 ----------------
// q at col h*64, k at 768+h*64, v at 1536+h*64.  grid(16,12,8), 4 waves.
// K tiles DMAd by the Tensor Data Mover (double-buffered), V staged transposed.
__global__ __launch_bounds__(128) void attn_kernel(
    const __bf16* __restrict__ qkv, __bf16* __restrict__ attn_out) {
  __shared__ alignas(16) __bf16 Ks[2][64][72];   // [buf][key][d], TDM-padded rows
  __shared__ alignas(16) __bf16 Vt[64][72];      // Vt[d][kn]
  __shared__ alignas(16) __bf16 Pl[4][16][72];   // per-wave P tile

  const int tid  = threadIdx.x;
  const int lane = tid & 31, wave = tid >> 5;
  const int g = lane >> 4, m = lane & 15;
  const int b = blockIdx.z, h = blockIdx.y, q0 = blockIdx.x * 64;
  const size_t S = 2304;
  const float scale = 0.125f;          // hd^-0.5, hd=64

  // preload Q fragments (A layout): this wave's 16 query rows, d = 0..63
  v16bf qa[2];
  {
    const __bf16* qrow = qkv + (size_t)(b*1024 + q0 + wave*16 + m) * S + h*64;
#pragma unroll
    for (int dk = 0; dk < 2; ++dk)
      qa[dk] = cat16(*(const bf8*)(qrow + dk*32 + g*8),
                     *(const bf8*)(qrow + dk*32 + 16 + g*8));
  }

  // kick off TDM for K tile 0
  if (wave == 0) {
    tdm_load_tile64x64_bf16(qkv + (size_t)(b*1024) * S + 768 + h*64,
                            (unsigned)(uintptr_t)&Ks[0][0][0], 2304u);
  }

  v8f o[4] = {};
  float rmax[8], rsum[8];
#pragma unroll
  for (int r = 0; r < 8; ++r) { rmax[r] = -__builtin_inff(); rsum[r] = 0.f; }

  for (int t = 0; t < 16; ++t) {
    const int k0 = t * 64;
    const int buf = t & 1;

    // ---- stage V tile transposed into LDS (overlaps the K DMA) ----
    {
      const int kn = tid >> 1, db = (tid & 1) * 32;
      const __bf16* vrow = qkv + (size_t)(b*1024 + k0 + kn) * S + 1536 + h*64 + db;
#pragma unroll
      for (int c = 0; c < 4; ++c) {
        bf8 v = *(const bf8*)(vrow + c*8);
#pragma unroll
        for (int j = 0; j < 8; ++j) Vt[db + c*8 + j][kn] = v[j];
      }
    }

    // ---- issue next K DMA, then guarantee tile t is resident ----
    if (wave == 0) {
      if (t < 15) {
        tdm_load_tile64x64_bf16(qkv + (size_t)(b*1024 + (t+1)*64) * S + 768 + h*64,
                                (unsigned)(uintptr_t)&Ks[buf ^ 1][0][0], 2304u);
        __builtin_amdgcn_s_wait_tensorcnt(1);   // tile t done, tile t+1 in flight
      } else {
        __builtin_amdgcn_s_wait_tensorcnt(0);
      }
    }
    __syncthreads();

    // ---- S = Q @ K^T (B frag = contiguous row of K tile in LDS) ----
    v8f s[4] = {};
#pragma unroll
    for (int nk = 0; nk < 4; ++nk) {
      const __bf16* krow = &Ks[buf][nk*16 + m][0];
#pragma unroll
      for (int dk = 0; dk < 2; ++dk) {
        v16bf kb = cat16(*(const bf8*)(krow + dk*32 + g*8),
                         *(const bf8*)(krow + dk*32 + 16 + g*8));
        s[nk] = WMMA_BF16(qa[dk], kb, s[nk]);
      }
    }

    // ---- online softmax (rows live in 16-lane halves) ----
#pragma unroll
    for (int r = 0; r < 8; ++r) {
      float mx = rowmax16(fmaxf(fmaxf(s[0][r], s[1][r]), fmaxf(s[2][r], s[3][r])));
      mx *= scale;
      const float nm    = fmaxf(rmax[r], mx);
      const float alpha = __expf(rmax[r] - nm);
      rmax[r] = nm;
      float ps = 0.f;
#pragma unroll
      for (int nk = 0; nk < 4; ++nk) {
        const float p = __expf(s[nk][r] * scale - nm);
        s[nk][r] = p;
        ps += p;
      }
      ps = rowsum16(ps);
      rsum[r] = rsum[r] * alpha + ps;
#pragma unroll
      for (int ni = 0; ni < 4; ++ni) o[ni][r] *= alpha;
    }

    // ---- P (C layout) -> per-wave LDS -> A layout ----
#pragma unroll
    for (int nk = 0; nk < 4; ++nk)
#pragma unroll
      for (int r = 0; r < 8; ++r)
        Pl[wave][g*8 + r][nk*16 + m] = (__bf16)s[nk][r];

    // ---- O += P @ V ----
#pragma unroll
    for (int kk = 0; kk < 2; ++kk) {
      const __bf16* prow = &Pl[wave][m][kk*32];
      v16bf pa = cat16(*(const bf8*)(prow + g*8), *(const bf8*)(prow + 16 + g*8));
#pragma unroll
      for (int ni = 0; ni < 4; ++ni) {
        const __bf16* vtr = &Vt[ni*16 + m][kk*32];
        v16bf vb = cat16(*(const bf8*)(vtr + g*8), *(const bf8*)(vtr + 16 + g*8));
        o[ni] = WMMA_BF16(pa, vb, o[ni]);
      }
    }
    __syncthreads();   // protect Vt (and Ks[buf]) before next iteration
  }

  // ---- epilogue: normalize, store bf16 to attn_out [8192][768] ----
#pragma unroll
  for (int r = 0; r < 8; ++r) {
    const float inv = 1.0f / rsum[r];
    const int qrow = q0 + wave*16 + g*8 + r;
    __bf16* orow = attn_out + (size_t)(b*1024 + qrow) * 768 + h*64;
#pragma unroll
    for (int ni = 0; ni < 4; ++ni)
      orow[ni*16 + m] = (__bf16)(o[ni][r] * inv);
  }
}

// ---------------- launch ----------------
extern "C" void kernel_launch(void* const* d_in, const int* in_sizes, int n_in,
                              void* d_out, int out_size, void* d_ws, size_t ws_size,
                              hipStream_t stream) {
  (void)in_sizes; (void)n_in; (void)out_size; (void)ws_size;
  const float* x      = (const float*)d_in[0];   // [8,1024,768]
  const float* qkv_w  = (const float*)d_in[1];   // [768,2304]
  const float* qkv_b  = (const float*)d_in[2];   // [2304]
  const float* proj_w = (const float*)d_in[3];   // [768,768]
  const float* proj_b = (const float*)d_in[4];   // [768]
  float* out = (float*)d_out;                    // [8,1024,768]

  __bf16* qkv  = (__bf16*)d_ws;                  // [8192][2304]
  __bf16* attn = qkv + (size_t)8192 * 2304;      // [8192][768]

  const int M = 8192, K = 768, N1 = 2304, N2 = 768;

  gemm_bias_kernel<float, true>
      <<<dim3(N1/128, M/128), dim3(256), 0, stream>>>(x, qkv_w, qkv_b, (void*)qkv, M, N1, K);

  attn_kernel<<<dim3(16, 12, 8), dim3(128), 0, stream>>>(qkv, attn);

  gemm_bias_kernel<__bf16, false>
      <<<dim3(768/128, M/128), dim3(256), 0, stream>>>(attn, proj_w, proj_b, (void*)out, M, N2, K);
}
